// AxialTimeAttention_3788161155765
// MI455X (gfx1250) — compile-verified, matrix operand order checked by reading, not run on hardware
//
#include <hip/hip_runtime.h>
#include <math.h>

// ---------------------------------------------------------------------------
// AxialTimeAttention for MI455X (gfx1250), wave32 + WMMA f32_16x16x32_f16,
// TDM tensor_load_to_lds for attention q/k/v staging.
// T=16 B=2 C=768 H=16 W=16 D=4, HEADS=12 HD=64, S = H*W*D = 1024,
// samples (t,b) = 32, spatial positions N = B*S = 2048.
// ---------------------------------------------------------------------------

typedef __attribute__((ext_vector_type(16))) _Float16     v16h;
typedef __attribute__((ext_vector_type(8)))  _Float16     v8h;
typedef __attribute__((ext_vector_type(8)))  float        v8f;
typedef __attribute__((ext_vector_type(4)))  unsigned int u32x4;
typedef __attribute__((ext_vector_type(8)))  unsigned int u32x8;

#define CC      768
#define SS      1024
#define NSMP    32          // T*B
#define HEADS   12
#define HD      64
#define TT      16
#define LDSTR   40          // LDS row stride in halves (80B, keeps 16B align)

// ---------------------------------------------------------------------------
// Kernel 1: GroupNorm statistics.  One block per (sample, head) = 384 blocks.
// Each group is 64 channels x 1024 spatial = 65536 contiguous floats.
// Emits per-channel scale = rstd*gn_w[c], shift = gn_b[c] - mean*scale.
// ---------------------------------------------------------------------------
__global__ void gn_stats_kernel(const float* __restrict__ x,
                                const float* __restrict__ gn_w,
                                const float* __restrict__ gn_b,
                                float* __restrict__ ws_scale,
                                float* __restrict__ ws_shift) {
  __shared__ float s_sum[256];
  __shared__ float s_sq[256];
  int blk  = blockIdx.x;
  int smp  = blk / HEADS;
  int head = blk - smp * HEADS;
  int tid  = threadIdx.x;

  const float* base = x + ((size_t)smp * CC + head * HD) * SS;
  float acc = 0.f, acc2 = 0.f;
  const float4* b4 = (const float4*)base;  // 65536 floats = 16384 float4
  for (int i = tid; i < 16384; i += 256) {
    float4 v = b4[i];
    acc  += v.x + v.y + v.z + v.w;
    acc2 += v.x * v.x + v.y * v.y + v.z * v.z + v.w * v.w;
  }
  s_sum[tid] = acc;
  s_sq[tid]  = acc2;
  __syncthreads();
  for (int off = 128; off > 0; off >>= 1) {
    if (tid < off) {
      s_sum[tid] += s_sum[tid + off];
      s_sq[tid]  += s_sq[tid + off];
    }
    __syncthreads();
  }
  float mean = s_sum[0] * (1.f / 65536.f);
  float var  = s_sq[0] * (1.f / 65536.f) - mean * mean;
  float rstd = rsqrtf(var + 1e-5f);
  if (tid < HD) {
    int c = head * HD + tid;
    float sc = rstd * gn_w[c];
    ws_scale[smp * CC + c] = sc;
    ws_shift[smp * CC + c] = gn_b[c] - mean * sc;
  }
}

// ---------------------------------------------------------------------------
// Fragment loader: WMMA 16-bit A (MxK) and B (KxN) fragments both use lane&15
// as the matrix row (A: M-row, B: N-column), with the K halves split by
// lane<16 -> K{k..k+7, k+16..k+23}, lane>=16 -> K{k+8..k+15, k+24..k+31}.
// ---------------------------------------------------------------------------
__device__ inline v16h ld_frag(const _Float16* rowp, int koff) {
  v8h lo = *(const v8h*)(rowp + koff);
  v8h hi = *(const v8h*)(rowp + koff + 16);
  v16h r;
#pragma unroll
  for (int i = 0; i < 8; i++) { r[i] = lo[i]; r[8 + i] = hi[i]; }
  return r;
}

__device__ inline v8f wmma_f16(v16h a, v16h b, v8f c) {
  return __builtin_amdgcn_wmma_f32_16x16x32_f16(false, a, false, b,
                                                (short)0, c, false, false);
}

// ---------------------------------------------------------------------------
// Tiled GEMM  D[o][s] = sum_c A[o][c] * B[c][s]  per sample.
// MODE 0: QKV projection (A = w_qkv f32, B = GroupNorm(x) fused on the fly;
//         epilogue scatters f16 into ws_qkv[npos][head][part][t][hd]).
// MODE 1: output projection (A = w_out, B = ws_o f16; epilogue fuses +b_out,
//         *gamma, +residual x, writes fp32 d_out).
// Block tile 128(M) x 128(N), 8 waves of 64x32, K-step 32, ping-pong LDS.
// ---------------------------------------------------------------------------
template <int MODE>
__device__ inline void stage_tiles(_Float16* aL, _Float16* bL,
                                   const float* __restrict__ Aw,
                                   const float* __restrict__ xin,
                                   const float* __restrict__ ws_scale,
                                   const float* __restrict__ ws_shift,
                                   const _Float16* __restrict__ Bf16,
                                   int smp, int m0, int n_blk, int kb,
                                   int tid) {
  // ---- A tile: 128 rows(o) x 32 k, f32 -> f16 -----------------------------
  {
    int r  = tid >> 1;
    int k2 = (tid & 1) * 16;
    const float* rowp = Aw + (size_t)(m0 + r) * CC + kb + k2;
    const float4* src = (const float4*)rowp;
    if (kb + 64 < CC) __builtin_prefetch(rowp + 64, 0, 3);  // next-next tile
    _Float16 tmp[16];
#pragma unroll
    for (int j = 0; j < 4; j++) {
      float4 v = src[j];
      tmp[4 * j + 0] = (_Float16)v.x;
      tmp[4 * j + 1] = (_Float16)v.y;
      tmp[4 * j + 2] = (_Float16)v.z;
      tmp[4 * j + 3] = (_Float16)v.w;
    }
    v8h* dst = (v8h*)(aL + r * LDSTR + k2);
    v8h t0, t1;
#pragma unroll
    for (int j = 0; j < 8; j++) { t0[j] = tmp[j]; t1[j] = tmp[8 + j]; }
    dst[0] = t0;
    dst[1] = t1;
  }
  // ---- B tile transposed: bT[s][k], 128 s x 32 k --------------------------
  {
    int kk = tid >> 3;           // 0..31
    int s0 = (tid & 7) * 16;     // 0..112
    int c  = kb + kk;
    if (MODE == 0) {
      float sc = ws_scale[smp * CC + c];
      float sh = ws_shift[smp * CC + c];
      const float* colp = xin + ((size_t)smp * CC + c) * SS + n_blk + s0;
      if (kb + 64 < CC) __builtin_prefetch(colp + 64 * SS, 0, 3);
      const float4* src = (const float4*)colp;
#pragma unroll
      for (int j = 0; j < 4; j++) {
        float4 v = src[j];
        bL[(s0 + 4 * j + 0) * LDSTR + kk] = (_Float16)(v.x * sc + sh);
        bL[(s0 + 4 * j + 1) * LDSTR + kk] = (_Float16)(v.y * sc + sh);
        bL[(s0 + 4 * j + 2) * LDSTR + kk] = (_Float16)(v.z * sc + sh);
        bL[(s0 + 4 * j + 3) * LDSTR + kk] = (_Float16)(v.w * sc + sh);
      }
    } else {
      const _Float16* colp = Bf16 + ((size_t)smp * CC + c) * SS + n_blk + s0;
      if (kb + 64 < CC) __builtin_prefetch(colp + 64 * SS, 0, 3);
      const v8h* src = (const v8h*)colp;
      v8h v0 = src[0], v1 = src[1];
#pragma unroll
      for (int i = 0; i < 8; i++) {
        bL[(s0 + i) * LDSTR + kk]     = v0[i];
        bL[(s0 + 8 + i) * LDSTR + kk] = v1[i];
      }
    }
  }
}

__device__ inline void do_compute(const _Float16* aL, const _Float16* bL,
                                  int wm, int wn, int row, int koff,
                                  v8f (&acc)[4][2]) {
  v16h a[4], b[2];
#pragma unroll
  for (int s = 0; s < 4; s++)
    a[s] = ld_frag(aL + (wm + s * 16 + row) * LDSTR, koff);
#pragma unroll
  for (int t = 0; t < 2; t++)
    b[t] = ld_frag(bL + (wn + t * 16 + row) * LDSTR, koff);
#pragma unroll
  for (int s = 0; s < 4; s++)
#pragma unroll
    for (int t = 0; t < 2; t++) acc[s][t] = wmma_f16(a[s], b[t], acc[s][t]);
}

template <int MODE>
__global__ void gemm_kernel(const float* __restrict__ Aw,
                            const float* __restrict__ bias,
                            const float* __restrict__ xin,       // MODE0 B src
                            const float* __restrict__ ws_scale,  // MODE0
                            const float* __restrict__ ws_shift,  // MODE0
                            const _Float16* __restrict__ Bf16,   // MODE1 B src
                            _Float16* __restrict__ qkv_out,      // MODE0 dst
                            const float* __restrict__ gamma,     // MODE1
                            const float* __restrict__ x_res,     // MODE1
                            float* __restrict__ out) {           // MODE1 dst
  __shared__ _Float16 a_lds[2][128 * LDSTR];
  __shared__ _Float16 b_lds[2][128 * LDSTR];

  const int m0    = blockIdx.x * 128;
  const int n_blk = blockIdx.y * 128;
  const int smp   = blockIdx.z;
  const int tid   = threadIdx.x;
  const int lane  = tid & 31;
  const int wv    = tid >> 5;
  const int wm    = (wv & 1) * 64;    // wave M offset (2 waves over 128)
  const int wn    = (wv >> 1) * 32;   // wave N offset (4 waves over 128)
  const int row   = lane & 15;
  const int koff  = (lane < 16) ? 0 : 8;

  v8f acc[4][2];
#pragma unroll
  for (int i = 0; i < 4; i++)
#pragma unroll
    for (int j = 0; j < 2; j++) acc[i][j] = (v8f)(0.f);

  stage_tiles<MODE>(a_lds[0], b_lds[0], Aw, xin, ws_scale, ws_shift, Bf16,
                    smp, m0, n_blk, 0, tid);

  // ping-pong over K: CC = 768 = 12 * 64, two 32-wide steps per iteration
  for (int kb = 0; kb < CC; kb += 64) {
    __syncthreads();  // buffer 0 (tile at kb) visible
    stage_tiles<MODE>(a_lds[1], b_lds[1], Aw, xin, ws_scale, ws_shift, Bf16,
                      smp, m0, n_blk, kb + 32, tid);
    do_compute(a_lds[0], b_lds[0], wm, wn, row, koff, acc);
    __syncthreads();  // buffer 1 (tile at kb+32) visible
    if (kb + 64 < CC)
      stage_tiles<MODE>(a_lds[0], b_lds[0], Aw, xin, ws_scale, ws_shift, Bf16,
                        smp, m0, n_blk, kb + 64, tid);
    do_compute(a_lds[1], b_lds[1], wm, wn, row, koff, acc);
  }

  // ---- epilogue -----------------------------------------------------------
  const int t_ = smp >> 1;   // time index
  const int b_ = smp & 1;    // batch index
#pragma unroll
  for (int ms = 0; ms < 4; ms++) {
#pragma unroll
    for (int ns = 0; ns < 2; ns++) {
#pragma unroll
      for (int r8 = 0; r8 < 8; r8++) {
        int o  = m0 + wm + ms * 16 + ((lane < 16) ? r8 : 8 + r8);
        int sl = n_blk + wn + ns * 16 + (lane & 15);
        float v = acc[ms][ns][r8] + bias[o];
        if (MODE == 0) {
          int head = o / 192;
          int rem  = o - head * 192;
          int part = rem / HD;        // 0=q 1=k 2=v
          int hd   = rem - part * HD;
          int npos = b_ * SS + sl;
          size_t idx =
              ((((size_t)npos * HEADS + head) * 3 + part) * TT + t_) * HD + hd;
          qkv_out[idx] = (_Float16)v;
        } else {
          size_t idx = ((size_t)smp * CC + o) * SS + sl;
          out[idx] = v * gamma[o] + x_res[idx];
        }
      }
    }
  }
}

// ---------------------------------------------------------------------------
// Kernel 3: attention.  One wave per (npos, head) task; 8 waves / block.
// q/k/v (6144 contiguous bytes) staged to LDS by the Tensor Data Mover with
// a 1-D descriptor; completion via s_wait_tensorcnt.  Computes S^T = K.Q^T
// via WMMA so the post-softmax probabilities land exactly in the A-fragment
// layout (lane&15 = query t) for O = P.V -- no LDS transpose needed.
// ---------------------------------------------------------------------------
__device__ inline float rel_bias(int t, int s, int head,
                                 const float* __restrict__ rel_emb) {
  int n   = t - s;                 // n = -rel, rel = s - t
  int ret = (n < 0) ? 16 : 0;      // NUM_BUCKETS/2
  int an  = n < 0 ? -n : n;
  int am  = an < 1 ? 1 : an;
  // max_exact = 8, log(MAX_DIST/max_exact) = log(16)
  int vl = 8 + (int)(logf((float)am * 0.125f) * (1.0f / logf(16.f)) * 8.f);
  vl = vl > 15 ? 15 : vl;
  int bucket = ret + ((an < 8) ? an : vl);
  return rel_emb[bucket * HEADS + head];
}

__global__ void attn_kernel(const _Float16* __restrict__ qkv,
                            const float* __restrict__ qn_w,
                            const float* __restrict__ qn_b,
                            const float* __restrict__ kn_w,
                            const float* __restrict__ kn_b,
                            const float* __restrict__ rel_emb,
                            _Float16* __restrict__ o_ws) {
  __shared__ _Float16 sm[8][3][TT * HD];  // per-wave q,k,v tiles (48 KB)
  const int wv   = __builtin_amdgcn_readfirstlane(threadIdx.x >> 5);
  const int lane = threadIdx.x & 31;
  const int task = blockIdx.x * 8 + wv;
  const int npos = task / HEADS;
  const int head = task - npos * HEADS;

  const _Float16* base = qkv + ((size_t)(npos * HEADS + head) * 3) * (TT * HD);
  _Float16* q_l = sm[wv][0];
  _Float16* k_l = sm[wv][1];
  _Float16* v_l = sm[wv][2];

  // ---- TDM: 1-D async copy of 3072 f16 (q|k|v) into this wave's LDS tile --
  {
    unsigned lds_off = (unsigned)(size_t)(void*)q_l;        // LDS byte offset
    unsigned long long ga = (unsigned long long)(size_t)base;
    u32x4 g0;
    g0[0] = 1u;                                             // count = 1
    g0[1] = lds_off;                                        // lds_addr
    g0[2] = (unsigned)ga;                                   // global_addr lo
    g0[3] = (unsigned)((ga >> 32) & 0x01FFFFFFu)            // global_addr hi
            | 0x80000000u;                                  // type = 2
    u32x8 g1;
    g1[0] = 0x00010000u;       // data_size = 1 (2 bytes)
    g1[1] = 0x0C000000u;       // tensor_dim0 low16 = 3072 in bits 63:48
    g1[2] = 0x00010000u;       // tensor_dim1 = 1 in bits 95:80
    g1[3] = 0x0C000000u;       // tile_dim0 = 3072 in bits 127:112
    g1[4] = 0u;                // tile_dim1/2 = 0 (unused)
    g1[5] = 3072u;             // tensor_dim0_stride = 3072
    g1[6] = 0u;
    g1[7] = 0u;
    asm volatile("tensor_load_to_lds %0, %1" :: "s"(g0), "s"(g1) : "memory");
    __builtin_amdgcn_s_wait_tensorcnt(0);
  }
  __syncthreads();

  // per-lane LayerNorm: lanes 0-15 -> q rows, lanes 16-31 -> k rows
  {
    int part = lane >> 4;
    int r    = lane & 15;
    _Float16* rp = sm[wv][part] + r * HD;
    const float* lw = part ? kn_w : qn_w;
    const float* lb = part ? kn_b : qn_b;
    float s1 = 0.f, s2 = 0.f;
#pragma unroll 8
    for (int i = 0; i < HD; i++) {
      float v = (float)rp[i];
      s1 += v;
      s2 += v * v;
    }
    float mean = s1 * (1.f / HD);
    float var  = s2 * (1.f / HD) - mean * mean;
    float rs   = rsqrtf(var + 1e-5f);
#pragma unroll 8
    for (int i = 0; i < HD; i++)
      rp[i] = (_Float16)(((float)rp[i] - mean) * rs * lw[i] + lb[i]);
  }
  __syncthreads();

  const int row  = lane & 15;
  const int koff = (lane < 16) ? 0 : 8;

  // S^T = K . Q^T  (M = key s, N = query t), K-dim = HD = 64 -> 2 WMMAs
  v8f accS = (v8f)(0.f);
  {
    v16h af0 = ld_frag(k_l + row * HD, koff);
    v16h bf0 = ld_frag(q_l + row * HD, koff);
    accS = wmma_f16(af0, bf0, accS);
    v16h af1 = ld_frag(k_l + row * HD + 32, koff);
    v16h bf1 = ld_frag(q_l + row * HD + 32, koff);
    accS = wmma_f16(af1, bf1, accS);
  }

  // softmax over s for each query t (t = lane&15; s split across wave halves)
  const int tq = lane & 15;
  float p[8];
  float mx = -1e30f;
#pragma unroll
  for (int r8 = 0; r8 < 8; r8++) {
    int sk = r8 + ((lane < 16) ? 0 : 8);
    float sc = accS[r8] * 0.125f + rel_bias(tq, sk, head, rel_emb);
    p[r8] = sc;
    mx = fmaxf(mx, sc);
  }
  mx = fmaxf(mx, __shfl_xor(mx, 16, 32));
  float sum = 0.f;
#pragma unroll
  for (int r8 = 0; r8 < 8; r8++) {
    p[r8] = __expf(p[r8] - mx);
    sum += p[r8];
  }
  sum += __shfl_xor(sum, 16, 32);
  float inv = 1.f / sum;

  // P already matches the A-fragment layout for O = P.V (K halves per lane)
  v16h pa;
#pragma unroll
  for (int r8 = 0; r8 < 8; r8++) pa[r8] = (_Float16)(p[r8] * inv);
#pragma unroll
  for (int r8 = 8; r8 < 16; r8++) pa[r8] = (_Float16)0.f;

  const int bpos = npos >> 10;
  const int spos = npos & (SS - 1);

#pragma unroll
  for (int ct = 0; ct < 4; ct++) {
    int cc    = ct * 16 + (lane & 15);
    int sbase = (lane < 16) ? 0 : 8;
    v16h vb;
#pragma unroll
    for (int i = 0; i < 8; i++) vb[i] = v_l[(sbase + i) * HD + cc];
#pragma unroll
    for (int i = 8; i < 16; i++) vb[i] = (_Float16)0.f;
    v8f accO = (v8f)(0.f);
    accO = wmma_f16(pa, vb, accO);
#pragma unroll
    for (int r8 = 0; r8 < 8; r8++) {
      int t   = (lane < 16) ? r8 : 8 + r8;
      int c   = head * HD + ct * 16 + (lane & 15);
      int smp = t * 2 + bpos;
      size_t idx = ((size_t)smp * CC + c) * SS + spos;
      o_ws[idx] = (_Float16)accO[r8];
    }
  }
}

// ---------------------------------------------------------------------------
// Host-side launch.  Workspace layout:
//   [0, 96K)        ws_scale  (32*768 f32)
//   [96K, 192K)     ws_shift
//   [192K, +151MB)  ws_qkv    (2048*12*3*16*64 f16)
//   then            ws_o      (32*768*1024 f16)
// ---------------------------------------------------------------------------
extern "C" void kernel_launch(void* const* d_in, const int* in_sizes, int n_in,
                              void* d_out, int out_size, void* d_ws,
                              size_t ws_size, hipStream_t stream) {
  const float* x       = (const float*)d_in[0];
  const float* gn_w    = (const float*)d_in[1];
  const float* gn_b    = (const float*)d_in[2];
  const float* w_qkv   = (const float*)d_in[3];
  const float* b_qkv   = (const float*)d_in[4];
  const float* qn_w    = (const float*)d_in[5];
  const float* qn_b    = (const float*)d_in[6];
  const float* kn_w    = (const float*)d_in[7];
  const float* kn_b    = (const float*)d_in[8];
  const float* rel_emb = (const float*)d_in[9];
  const float* w_out   = (const float*)d_in[10];
  const float* b_out   = (const float*)d_in[11];
  const float* gamma   = (const float*)d_in[12];
  float* out = (float*)d_out;

  char* ws = (char*)d_ws;
  float*     ws_scale = (float*)ws;
  float*     ws_shift = (float*)(ws + 98304);
  _Float16*  ws_qkv   = (_Float16*)(ws + 196608);
  _Float16*  ws_o     = (_Float16*)(ws + 196608 + (size_t)150994944);

  // 1) GroupNorm stats -> per-channel scale/shift
  gn_stats_kernel<<<dim3(NSMP * HEADS), dim3(256), 0, stream>>>(
      x, gn_w, gn_b, ws_scale, ws_shift);

  // 2) QKV GEMM (2304 x 1024 x 768 per sample), GN fused into B staging
  gemm_kernel<0><<<dim3(18, 8, NSMP), dim3(256), 0, stream>>>(
      w_qkv, b_qkv, x, ws_scale, ws_shift, (const _Float16*)nullptr, ws_qkv,
      (const float*)nullptr, (const float*)nullptr, (float*)nullptr);

  // 3) attention: 2048*12 wave-tasks, 8 per block
  attn_kernel<<<dim3(3072), dim3(256), 0, stream>>>(
      ws_qkv, qn_w, qn_b, kn_w, kn_b, rel_emb, ws_o);

  // 4) output projection + gamma scale + residual
  gemm_kernel<1><<<dim3(6, 8, NSMP), dim3(256), 0, stream>>>(
      w_out, b_out, (const float*)nullptr, (const float*)nullptr,
      (const float*)nullptr, ws_o, (_Float16*)nullptr, gamma, x, out);

  (void)in_sizes; (void)n_in; (void)out_size; (void)ws_size;
}
